// ContextualAttention_10462540333340
// MI455X (gfx1250) — matrix operand order repaired and checked by reading, not compiled, not run on hardware
//
#include <hip/hip_runtime.h>
#include <hip/hip_bf16.h>
#include <math.h>

typedef __attribute__((ext_vector_type(16))) _Float16 v16h;
typedef __attribute__((ext_vector_type(8)))  float    v8f;
typedef int i128v __attribute__((vector_size(16)));

#define C_CH   128
#define HFULL  128
#define HD     64
#define P_POS  4096   // HD*HD
#define K1     1152   // C*3*3
#define N2     2048   // C*4*4

// workspace layout (bytes), reused per sample
#define FH_OFF 0u
#define WN_OFF 9437184u                 // 4096*1152*2
#define BR_OFF (WN_OFF + 9437184u)      // 2048*4096*2
#define S_OFF  (BR_OFF + 16777216u)     // 4096*4096*4  (reused as P: 4096*2048*4)
#define YH_OFF (S_OFF + 67108864u)      // 4096*4096*2
#define MM_OFF (YH_OFF + 33554432u)     // 4096*4

#define ASYNC_LDS_OK (__has_builtin(__builtin_amdgcn_global_load_async_to_lds_b128))

__device__ __forceinline__ void async_b128(const void* g, void* l) {
#if ASYNC_LDS_OK
    __builtin_amdgcn_global_load_async_to_lds_b128(
        (__attribute__((address_space(1))) i128v*)(g),
        (__attribute__((address_space(3))) i128v*)(l), 0, 0);
#endif
}

__device__ __forceinline__ void wait_async0() {
#if __has_builtin(__builtin_amdgcn_s_wait_asynccnt)
    __builtin_amdgcn_s_wait_asynccnt(0);
#else
    asm volatile("s_wait_asynccnt 0x0" ::: "memory");
#endif
}

// ---------------- im2col: Fh[p][k] = f_d patch, k = c*9 + kh*3 + kw ----------------
__global__ __launch_bounds__(256)
void build_F(const float* __restrict__ f, _Float16* __restrict__ Fh, int s) {
    int idx = blockIdx.x * 256 + threadIdx.x;
    if (idx >= P_POS * K1) return;
    int p = idx / K1, k = idx - p * K1;
    int c = k / 9, r = k - c * 9;
    int kh = r / 3, kw = r - kh * 3;
    int y = p >> 6, x = p & 63;
    int yy = y + kh - 1, xx = x + kw - 1;
    float v = 0.f;
    if (yy >= 0 && yy < HD && xx >= 0 && xx < HD)
        v = f[(((size_t)s * C_CH + c) * HFULL + 2 * yy) * HFULL + 2 * xx];
    Fh[idx] = (_Float16)v;
}

// ------------- normalized patches of b_d: Wn[q][k], norm over 1152-elem patch -------------
__global__ __launch_bounds__(256)
void build_Wn(const float* __restrict__ b, _Float16* __restrict__ Wn, int s) {
    __shared__ float pv[K1];
    __shared__ float part[256];
    int t = threadIdx.x, q = blockIdx.x;
    int yq = q >> 6, xq = q & 63;
    float ss = 0.f;
    for (int e = t; e < K1; e += 256) {
        int c = e / 9, r = e - c * 9;
        int kh = r / 3, kw = r - kh * 3;
        int yy = yq + kh - 1, xx = xq + kw - 1;
        float v = 0.f;
        if (yy >= 0 && yy < HD && xx >= 0 && xx < HD)
            v = b[(((size_t)s * C_CH + c) * HFULL + 2 * yy) * HFULL + 2 * xx];
        pv[e] = v;
        ss += v * v;
    }
    part[t] = ss;
    __syncthreads();
    for (int off = 128; off > 0; off >>= 1) {
        if (t < off) part[t] += part[t + off];
        __syncthreads();
    }
    float rn = 1.f / fmaxf(sqrtf(part[0]), 1e-4f);
    __syncthreads();
    for (int e = t; e < K1; e += 256)
        Wn[(size_t)q * K1 + e] = (_Float16)(pv[e] * rn);
}

// ------------- raw 4x4 patches transposed: BrT[n][q], n = c*16 + kh*4 + kw -------------
__global__ __launch_bounds__(256)
void build_Br(const float* __restrict__ b, _Float16* __restrict__ BrT, int s) {
    int idx = blockIdx.x * 256 + threadIdx.x;
    if (idx >= N2 * P_POS) return;
    int n = idx >> 12, q = idx & 4095;
    int c = n >> 4, r = n & 15;
    int kh = r >> 2, kw = r & 3;
    int yq = q >> 6, xq = q & 63;
    int Y = 2 * yq - 1 + kh, X = 2 * xq - 1 + kw;
    float v = 0.f;
    if (Y >= 0 && Y < HFULL && X >= 0 && X < HFULL)
        v = b[(((size_t)s * C_CH + c) * HFULL + Y) * HFULL + X];
    BrT[idx] = (_Float16)v;
}

// ------------- mask validity per patch: mm[q] = (mean of 3x3 mask_d patch == 0) -------------
__global__ __launch_bounds__(256)
void build_mm(const float* __restrict__ mask, float* __restrict__ mm, int s) {
    int q = blockIdx.x * 256 + threadIdx.x;
    if (q >= P_POS) return;
    int yq = q >> 6, xq = q & 63;
    float acc = 0.f;
    for (int kh = 0; kh < 3; ++kh)
        for (int kw = 0; kw < 3; ++kw) {
            int yy = yq + kh - 1, xx = xq + kw - 1;
            if (yy >= 0 && yy < HD && xx >= 0 && xx < HD)
                acc += mask[((size_t)s * 512 + 8 * yy) * 512 + 8 * xx];
        }
    mm[q] = (acc == 0.f) ? 1.f : 0.f;
}

// ------------- tiled WMMA GEMM: D[M][N] = A[M][K] * Bt[N][K]^T, f16 in / f32 out -------------
// BM=BN=128, BK=32; 256 threads = 8 waves in 2x4 grid; each wave: 64x32 = 4x2 WMMA tiles.
// Double-buffered LDS; tile k+1 staged via async global->LDS DMA while tile k runs WMMA.
__global__ __launch_bounds__(256)
void gemm_nt_f16(const _Float16* __restrict__ A, const _Float16* __restrict__ Bt,
                 float* __restrict__ D, int M, int N, int K) {
    __shared__ __align__(16) _Float16 lA[2][128 * 32];
    __shared__ __align__(16) _Float16 lB[2][128 * 32];

    const int tid  = threadIdx.x;
    const int lane = tid & 31;
    const int wid  = tid >> 5;
    const int wm   = wid >> 2;      // 0..1
    const int wn   = wid & 3;       // 0..3
    const int lh   = lane >> 4;     // lane-half
    const int l15  = lane & 15;
    const int bm   = blockIdx.y, bn = blockIdx.x;

    v8f zero = {};
    v8f acc[4][2];
#pragma unroll
    for (int i = 0; i < 4; ++i)
#pragma unroll
        for (int j = 0; j < 2; ++j) acc[i][j] = zero;

    const int nIter = K >> 5;

#if ASYNC_LDS_OK
    // ---------------- async-DMA double-buffered pipeline ----------------
    {
        uint4* dA = (uint4*)lA[0];
        uint4* dB = (uint4*)lB[0];
#pragma unroll
        for (int u = 0; u < 2; ++u) {
            int c = tid + u * 256, row = c >> 2, jc = c & 3;
            async_b128(A  + (size_t)(bm * 128 + row) * K + jc * 8, dA + c);
            async_b128(Bt + (size_t)(bn * 128 + row) * K + jc * 8, dB + c);
        }
    }
    for (int it = 0; it < nIter; ++it) {
        const int cur = it & 1;
        wait_async0();          // own stage(it) writes have landed in LDS
        __syncthreads();        // everyone's stage(it) landed; prev readers done
        if (it + 1 < nIter) {   // overlap stage(it+1) with WMMA of tile it
            const int kk = (it + 1) << 5;
            uint4* dA = (uint4*)lA[cur ^ 1];
            uint4* dB = (uint4*)lB[cur ^ 1];
#pragma unroll
            for (int u = 0; u < 2; ++u) {
                int c = tid + u * 256, row = c >> 2, jc = c & 3;
                async_b128(A  + (size_t)(bm * 128 + row) * K + kk + jc * 8, dA + c);
                async_b128(Bt + (size_t)(bn * 128 + row) * K + kk + jc * 8, dB + c);
            }
        }
        const uint4* lA4 = (const uint4*)lA[cur];
        const uint4* lB4 = (const uint4*)lB[cur];

        v16h af[4];
#pragma unroll
        for (int i = 0; i < 4; ++i) {
            int r = wm * 64 + i * 16 + l15;
            uint4* pa = (uint4*)&af[i];
            pa[0] = lA4[r * 4 + lh];
            pa[1] = lA4[r * 4 + 2 + lh];
        }
        v16h bf[2];
#pragma unroll
        for (int j = 0; j < 2; ++j) {
            int n = wn * 32 + j * 16 + l15;
            uint4* pb = (uint4*)&bf[j];
            pb[0] = lB4[n * 4 + lh];
            pb[1] = lB4[n * 4 + 2 + lh];
        }
#pragma unroll
        for (int i = 0; i < 4; ++i)
#pragma unroll
            for (int j = 0; j < 2; ++j)
                acc[i][j] = __builtin_amdgcn_wmma_f32_16x16x32_f16(
                    false, af[i], false, bf[j], (short)0, acc[i][j], false, false);
    }
#else
    // ---------------- fallback: batched register staging, single buffer ----------------
    uint4* lA4 = (uint4*)lA[0];
    uint4* lB4 = (uint4*)lB[0];
    for (int it = 0; it < nIter; ++it) {
        const int kk = it << 5;
        int r0 = tid >> 2, j0 = tid & 3;
        int c1 = tid + 256, r1 = c1 >> 2, j1 = c1 & 3;
        uint4 ra0 = *(const uint4*)(A  + (size_t)(bm * 128 + r0) * K + kk + j0 * 8);
        uint4 rb0 = *(const uint4*)(Bt + (size_t)(bn * 128 + r0) * K + kk + j0 * 8);
        uint4 ra1 = *(const uint4*)(A  + (size_t)(bm * 128 + r1) * K + kk + j1 * 8);
        uint4 rb1 = *(const uint4*)(Bt + (size_t)(bn * 128 + r1) * K + kk + j1 * 8);
        __syncthreads();
        lA4[tid] = ra0; lB4[tid] = rb0;
        lA4[c1]  = ra1; lB4[c1]  = rb1;
        __syncthreads();

        v16h af[4];
#pragma unroll
        for (int i = 0; i < 4; ++i) {
            int r = wm * 64 + i * 16 + l15;
            uint4* pa = (uint4*)&af[i];
            pa[0] = lA4[r * 4 + lh];
            pa[1] = lA4[r * 4 + 2 + lh];
        }
        v16h bf[2];
#pragma unroll
        for (int j = 0; j < 2; ++j) {
            int n = wn * 32 + j * 16 + l15;
            uint4* pb = (uint4*)&bf[j];
            pb[0] = lB4[n * 4 + lh];
            pb[1] = lB4[n * 4 + 2 + lh];
        }
#pragma unroll
        for (int i = 0; i < 4; ++i)
#pragma unroll
            for (int j = 0; j < 2; ++j)
                acc[i][j] = __builtin_amdgcn_wmma_f32_16x16x32_f16(
                    false, af[i], false, bf[j], (short)0, acc[i][j], false, false);
    }
#endif

    // C/D layout: VGPR e, lanes 0-15 -> M=e, N=lane; lanes 16-31 -> M=e+8, N=lane-16
#pragma unroll
    for (int i = 0; i < 4; ++i)
#pragma unroll
        for (int j = 0; j < 2; ++j)
#pragma unroll
            for (int e = 0; e < 8; ++e) {
                int gr = bm * 128 + wm * 64 + i * 16 + lh * 8 + e;
                int gc = bn * 128 + wn * 32 + j * 16 + l15;
                __builtin_nontemporal_store(acc[i][j][e], &D[(size_t)gr * N + gc]);
            }
}

// ------------- row softmax (scale=2550) + argmax + offsets; writes f16 probs -------------
__global__ __launch_bounds__(256)
void softmax_rows(const float* __restrict__ S, const float* __restrict__ mm,
                  _Float16* __restrict__ Yh, int* __restrict__ offs, int s) {
    __shared__ float red[256];
    __shared__ int  redi[256];
    const int t = threadIdx.x, p = blockIdx.x;
    const float scale = 2550.f;
    const size_t base = (size_t)p * P_POS;

    float lmax = -3.4e38f;
    for (int q = t; q < P_POS; q += 256)
        lmax = fmaxf(lmax, S[base + q] * mm[q]);
    red[t] = lmax; __syncthreads();
    for (int off = 128; off > 0; off >>= 1) {
        if (t < off) red[t] = fmaxf(red[t], red[t + off]);
        __syncthreads();
    }
    float vmax = red[0]; __syncthreads();

    float lsum = 0.f;
    for (int q = t; q < P_POS; q += 256)
        lsum += __expf(scale * (S[base + q] * mm[q] - vmax));
    red[t] = lsum; __syncthreads();
    for (int off = 128; off > 0; off >>= 1) {
        if (t < off) red[t] += red[t + off];
        __syncthreads();
    }
    float rden = 1.f / red[0]; __syncthreads();

    float bv = -1.f; int bi = P_POS;
    for (int q = t; q < P_POS; q += 256) {
        float pv = __expf(scale * (S[base + q] * mm[q] - vmax)) * rden * mm[q];
        Yh[base + q] = (_Float16)pv;
        if (pv > bv) { bv = pv; bi = q; }
    }
    red[t] = bv; redi[t] = bi; __syncthreads();
    for (int off = 128; off > 0; off >>= 1) {
        if (t < off) {
            if (red[t + off] > red[t] || (red[t + off] == red[t] && redi[t + off] < redi[t])) {
                red[t] = red[t + off]; redi[t] = redi[t + off];
            }
        }
        __syncthreads();
    }
    if (t == 0) {
        int off0 = redi[0];
        int y = p >> 6, x = p & 63;
        offs[(((size_t)s * 2 + 0) * HD + y) * HD + x] = (off0 >> 6) - y;
        offs[(((size_t)s * 2 + 1) * HD + y) * HD + x] = (off0 & 63) - x;
    }
}

// ------------- 4-tap stride-2 overlap-add (transposed conv gather), /4 -------------
__global__ __launch_bounds__(256)
void gather_out(const float* __restrict__ P, float* __restrict__ out, int s) {
    int idx = blockIdx.x * 256 + threadIdx.x;
    if (idx >= C_CH * HFULL * HFULL) return;
    int c = idx >> 14;
    int rem = idx & 16383;
    int Y = rem >> 7, X = rem & 127;
    float sum = 0.f;
    for (int kh = 0; kh < 4; ++kh) {
        int ty = Y + 1 - kh;
        if (ty < 0 || (ty & 1)) continue;
        int y = ty >> 1;
        if (y >= HD) continue;
        for (int kw = 0; kw < 4; ++kw) {
            int tx = X + 1 - kw;
            if (tx < 0 || (tx & 1)) continue;
            int x = tx >> 1;
            if (x >= HD) continue;
            sum += P[(size_t)(y * HD + x) * N2 + (c << 4) + (kh << 2) + kw];
        }
    }
    out[(((size_t)s * C_CH + c) * HFULL + Y) * HFULL + X] = 0.25f * sum;
}

extern "C" void kernel_launch(void* const* d_in, const int* in_sizes, int n_in,
                              void* d_out, int out_size, void* d_ws, size_t ws_size,
                              hipStream_t stream) {
    const float* f    = (const float*)d_in[0];
    const float* b    = (const float*)d_in[1];
    const float* mask = (const float*)d_in[2];
    float* out_y = (float*)d_out;
    int*   offs  = (int*)(out_y + (size_t)4 * C_CH * HFULL * HFULL);

    char* ws = (char*)d_ws;
    _Float16* Fh  = (_Float16*)(ws + FH_OFF);
    _Float16* Wn  = (_Float16*)(ws + WN_OFF);
    _Float16* BrT = (_Float16*)(ws + BR_OFF);
    float*    S   = (float*)(ws + S_OFF);     // also reused as P after softmax
    _Float16* Yh  = (_Float16*)(ws + YH_OFF);
    float*    mm  = (float*)(ws + MM_OFF);

    for (int s = 0; s < 4; ++s) {
        build_F <<<(P_POS * K1 + 255) / 256, 256, 0, stream>>>(f, Fh, s);
        build_Wn<<<P_POS, 256, 0, stream>>>(b, Wn, s);
        build_Br<<<(N2 * P_POS + 255) / 256, 256, 0, stream>>>(b, BrT, s);
        build_mm<<<(P_POS + 255) / 256, 256, 0, stream>>>(mask, mm, s);

        // S = Fh (4096x1152) * Wn^T  -> 4096x4096 scores
        gemm_nt_f16<<<dim3(P_POS / 128, P_POS / 128), 256, 0, stream>>>(Fh, Wn, S, P_POS, P_POS, K1);

        softmax_rows<<<P_POS, 256, 0, stream>>>(S, mm, Yh, offs, s);

        // P = Yh (4096x4096) * BrT^T -> 4096x2048 (reuse S buffer)
        gemm_nt_f16<<<dim3(N2 / 128, P_POS / 128), 256, 0, stream>>>(Yh, BrT, S, P_POS, N2, P_POS);

        gather_out<<<(C_CH * HFULL * HFULL + 255) / 256, 256, 0, stream>>>(S, out_y, s);
    }
}
